// FlexConvolutionTransposed_33835752358494
// MI455X (gfx1250) — compile-verified, matrix operand order checked by reading, not run on hardware
//
#include <hip/hip_runtime.h>

// ---------------------------------------------------------------------------
// FlexConvolutionTransposed for MI455X (gfx1250, wave32, WMMA)
//
//   out[b,o,j] = bias[o] + sum_{k,n : nbh[b,k,n]==j}
//                  ( sum_d (pos[b,d,j]-pos[b,d,n]) * (theta_d^T f)[b,o,n]
//                    + (wb^T f)[b,o,n] )
//
// One combined GEMM  W4[256x64] x features[64xN]  via v_wmma_f32_16x16x32_f16
// (f16 in, f32 accumulate).  Features are staged coalesced -> LDS in WMMA
// B-fragment layout (loaded from global exactly once per tile), results kept
// in padded LDS, then fused delta-combine + atomic scatter into the f32
// output (which is L2-resident: 16 MB << 192 MB).
// ---------------------------------------------------------------------------

typedef __attribute__((ext_vector_type(16))) _Float16 v16h;
typedef __attribute__((ext_vector_type(8)))  float    v8f;

static constexpr int BATCH    = 4;
static constexpr int CIN      = 64;
static constexpr int COUT     = 64;
static constexpr int NPTS     = 16384;
static constexpr int KNBH     = 8;

static constexpr int ROWS     = 4 * COUT;     // 256 combined output rows (3x theta + wb)
static constexpr int TILE_PTS = 32;           // points per workgroup
static constexpr int LDS_STR  = 33;           // padded row stride (dwords) -> conflict-free
static constexpr int NPAIRS   = TILE_PTS * KNBH;   // 256 (point, neighbor) pairs per block
static constexpr int A_ELEMS  = 16 /*m-tiles*/ * 2 /*k-chunks*/ * 32 /*lanes*/ * 16 /*halves*/;
static constexpr int B_HALVES = 2 /*n-tiles*/ * 2 /*k-chunks*/ * 32 /*lanes*/ * 16 /*halves*/;

// K index held by half h (0..15) for a lane in group g (0 or 1), chunk kc (0 or 1).
// Matches the CDNA5 ISA 16-bit A-matrix 16x32 layout (CDNA5 ISA 7.12.2).
__device__ __forceinline__ int kmap(int h, int g, int kc) {
    return 32 * kc + 16 * (h >> 3) + 8 * g + (h & 7);
}

// ---------------- kernel 1: out[b,o,n] = bias[o] (float4 stores) -----------
__global__ void init_out_kernel(const float* __restrict__ bias,
                                float* __restrict__ out) {
    size_t idx = (size_t)blockIdx.x * blockDim.x + threadIdx.x;   // over B*COUT*N/4
    size_t nwords = (size_t)BATCH * COUT * NPTS / 4;
    if (idx >= nwords) return;
    int o = (int)((idx / (NPTS / 4)) % COUT);
    float v = bias[o];
    ((float4*)out)[idx] = make_float4(v, v, v, v);
}

// ---------------- kernel 2: pack W4 = [theta0|theta1|theta2|wb] ------------
// into f16 A-matrix fragments in d_ws:  wsA[((mt*2+kc)*32+lane)*16 + h]
__global__ void pack_weights_kernel(const float* __restrict__ theta,   // [3,CIN,COUT]
                                    const float* __restrict__ wbias,   // [CIN,COUT]
                                    _Float16* __restrict__ wsA) {
    int e = blockIdx.x * blockDim.x + threadIdx.x;
    if (e >= A_ELEMS) return;
    int h    = e & 15;
    int lane = (e >> 4) & 31;
    int kc   = (e >> 9) & 1;
    int mt   = e >> 10;                  // 0..15
    int row  = mt * 16 + (lane & 15);    // combined output row 0..255
    int g    = lane >> 4;
    int Kk   = kmap(h, g, kc);           // cin index 0..63
    float w;
    if (row < 3 * COUT) {
        int d = row >> 6, o = row & 63;
        w = theta[((size_t)d * CIN + Kk) * COUT + o];
    } else {
        int o = row - 3 * COUT;
        w = wbias[(size_t)Kk * COUT + o];
    }
    wsA[e] = (_Float16)w;
}

// ---------------- kernel 3: fused WMMA GEMM + delta combine + scatter ------
__global__ void __launch_bounds__(256)
flexconv_kernel(const float* __restrict__ features,   // [B,CIN,N]
                const int*   __restrict__ nbh,        // [B,K,N]
                const float* __restrict__ pos,        // [B,3,N]
                const _Float16* __restrict__ wsA,     // packed A fragments
                float* __restrict__ out) {            // [B,COUT,N]
    __shared__ float ftfb[ROWS * LDS_STR];            // 33792 B: GEMM results
    __shared__ float pdx[NPAIRS], pdy[NPAIRS], pdz[NPAIRS];
    __shared__ int   pj[NPAIRS];                      // + 4 KB pair data
    __shared__ __align__(32) _Float16 bstage[B_HALVES]; // 4 KB B fragments

    const int tid        = threadIdx.x;
    const int blocksPerB = NPTS / TILE_PTS;           // 512
    const int b          = blockIdx.x / blocksPerB;
    const int n0         = (blockIdx.x % blocksPerB) * TILE_PTS;

    // --- stage per-(point,k) gather data: neighbor index + position delta ---
    {
        int q  = tid;                                  // exactly 256 pairs
        int pt = q >> 3, k = q & 7;
        int n  = n0 + pt;
        int j  = nbh[((size_t)b * KNBH + k) * NPTS + n];
        const float* posb = pos + (size_t)b * 3 * NPTS;
        pj[q]  = j;
        pdx[q] = posb[0 * NPTS + j] - posb[0 * NPTS + n];
        pdy[q] = posb[1 * NPTS + j] - posb[1 * NPTS + n];
        pdz[q] = posb[2 * NPTS + j] - posb[2 * NPTS + n];
    }

    // --- stage the 64(Cin) x 32(pt) feature tile into LDS, f16, B-fragment
    //     layout.  Global reads fully coalesced: per pass 8 Cin rows x 32
    //     consecutive floats; loaded exactly ONCE per block (was 4x before).
    {
        const int ncol = tid & 31;                    // 0..31 within tile
        const int krow = tid >> 5;                    // 0..7
        const float* featb = features + (size_t)b * CIN * NPTS + n0;
        const int nt   = ncol >> 4;
        const int lcol = ncol & 15;
        #pragma unroll
        for (int pass = 0; pass < 8; ++pass) {
            const int Kk  = krow + 8 * pass;          // 0..63
            const float v = featb[(size_t)Kk * NPTS + ncol];
            // inverse of kmap: K = 32*kc + 16*t1 + 8*g + k0  ->  h = 8*t1 + k0
            const int kc  = Kk >> 5;
            const int rem = Kk & 31;
            const int g   = (rem >> 3) & 1;
            const int h   = 8 * (rem >> 4) + (rem & 7);
            const int lane = g * 16 + lcol;
            bstage[(((nt * 2 + kc) * 32) + lane) * 16 + h] = (_Float16)v;
        }
    }
    __syncthreads();

    // --- GEMM phase: ftfb[row, pt] = sum_i W4[row,i] * feat[b,i,n0+pt] ------
    // 8 waves: wave -> (n-tile = wave&1, m-tiles = {wave>>1 + 4*i}).
    {
        const int wave  = tid >> 5;
        const int lane  = tid & 31;
        const int nt    = wave & 1;
        const int mbase = wave >> 1;
        const int g     = lane >> 4;
        const int lcol  = lane & 15;

        // B fragments now come from LDS (2 x 32B vector loads per wave lane)
        const v16h bfrag0 = *(const v16h*)(bstage + ((nt * 2 + 0) * 32 + lane) * 16);
        const v16h bfrag1 = *(const v16h*)(bstage + ((nt * 2 + 1) * 32 + lane) * 16);

        #pragma unroll
        for (int i = 0; i < 4; ++i) {
            const int mt = mbase + 4 * i;
            v8f c = {};
            {
                v16h a = *(const v16h*)(wsA + (((size_t)mt * 2 + 0) * 32 + lane) * 16);
                c = __builtin_amdgcn_wmma_f32_16x16x32_f16(
                        false, a, false, bfrag0, (short)0, c, false, false);
            }
            {
                v16h a = *(const v16h*)(wsA + (((size_t)mt * 2 + 1) * 32 + lane) * 16);
                c = __builtin_amdgcn_wmma_f32_16x16x32_f16(
                        false, a, false, bfrag1, (short)0, c, false, false);
            }
            // C/D layout: VGPR r, lane group g -> M = 8*g + r, N = lane&15
            const int colBase = nt * 16 + lcol;
            #pragma unroll
            for (int r = 0; r < 8; ++r) {
                int row = mt * 16 + 8 * g + r;
                ftfb[row * LDS_STR + colBase] = c[r];
            }
        }
    }
    __syncthreads();

    // --- scatter phase: 4 concurrent pairs, 64 output channels per pair ----
    {
        const int o = tid & 63;
        for (int pp = tid >> 6; pp < NPAIRS; pp += 4) {
            const int   pt = pp >> 3;
            const float dx = pdx[pp], dy = pdy[pp], dz = pdz[pp];
            const int   j  = pj[pp];
            // padded stride 33 -> 64 lanes with consecutive o hit 64 distinct banks
            const float c0 = ftfb[(0 * COUT + o) * LDS_STR + pt];
            const float c1 = ftfb[(1 * COUT + o) * LDS_STR + pt];
            const float c2 = ftfb[(2 * COUT + o) * LDS_STR + pt];
            const float cb = ftfb[(3 * COUT + o) * LDS_STR + pt];
            const float contrib = dx * c0 + dy * c1 + dz * c2 + cb;
            atomicAdd(&out[((size_t)b * COUT + o) * NPTS + j], contrib);
        }
    }
}

// ---------------------------------------------------------------------------
extern "C" void kernel_launch(void* const* d_in, const int* in_sizes, int n_in,
                              void* d_out, int out_size, void* d_ws, size_t ws_size,
                              hipStream_t stream) {
    const float* features = (const float*)d_in[0];  // [B,CIN,N]
    const float* theta    = (const float*)d_in[1];  // [3,CIN,COUT]
    const float* wbias    = (const float*)d_in[2];  // [CIN,COUT]
    const int*   nbh      = (const int*)d_in[3];    // [B,K,N]
    const float* pos      = (const float*)d_in[4];  // [B,3,N]
    const float* bias     = (const float*)d_in[5];  // [COUT]
    float*       out      = (float*)d_out;          // [B,COUT,N]
    _Float16*    wsA      = (_Float16*)d_ws;        // 32 KB packed weights

    {   // out = bias (needed every call: scatter accumulates with atomics)
        int nwords = BATCH * COUT * NPTS / 4;
        init_out_kernel<<<(nwords + 255) / 256, 256, 0, stream>>>(bias, out);
    }
    pack_weights_kernel<<<(A_ELEMS + 255) / 256, 256, 0, stream>>>(theta, wbias, wsA);
    flexconv_kernel<<<BATCH * (NPTS / TILE_PTS), 256, 0, stream>>>(
        features, nbh, pos, wsA, out);
}
